// Seq2SeqCoupletModel_49752901157211
// MI455X (gfx1250) — matrix operand order repaired and checked by reading, not compiled, not run on hardware
//
#include <hip/hip_runtime.h>
#include <hip/hip_bf16.h>
#include <stdint.h>

// ---------------------------------------------------------------------------
// CDNA5 (gfx1250) seq2seq forward. All GEMMs use v_wmma_f32_16x16x32_bf16.
// Operands are pre-packed into the ISA fragment layouts (CDNA5 ISA
// §7.12.2). GEMM is register-blocked 4x in M: one B-fragment load feeds four
// WMMAs (4x B reuse; weights are the large operand in the recurrent and output GEMMs).
// ---------------------------------------------------------------------------

typedef __attribute__((ext_vector_type(16))) __bf16 v16bf;
typedef __attribute__((ext_vector_type(8)))  float  v8f;

// ---- fragment index helpers (bf16 element index inside frag buffers) ------
// A-matrix 16x32 bf16 (ISA table): lanes 0-15: M=lane, elems 0..7 -> K=0..7,
// elems 8..15 -> K=16..23; lanes 16-31: M=lane-16, K ranges 8..15 / 24..31.
__device__ __forceinline__ size_t a_frag_index(int m, int k, int KC) {
  int mt = m >> 4, kt = k >> 5, kr = k & 31;
  int hi, e;
  if (kr < 16) { hi = (kr >> 3) & 1; e = kr & 7; }
  else         { int k2 = kr - 16; hi = (k2 >> 3) & 1; e = 8 + (k2 & 7); }
  int lane = (m & 15) | (hi << 4);
  return ((size_t)(mt * KC + kt) * 32 + lane) * 16 + e;
}
// B-matrix 32x16 bf16: same K order as A: lanes 0-15 elems 0..7 -> K=0..7,
// elems 8..15 -> K=16..23; lanes 16-31 K=8..15 / 24..31 (N = lane%16).
__device__ __forceinline__ size_t b_frag_index(int n, int k, int KC) {
  int nt = n >> 4, kt = k >> 5, kr = k & 31;
  int hi = (kr >> 3) & 1, e = (kr & 7) + ((kr >> 4) << 3);
  int lane = (n & 15) | (hi << 4);
  return ((size_t)(nt * KC + kt) * 32 + lane) * 16 + e;
}

__device__ __forceinline__ float sigm(float x) { return 1.0f / (1.0f + __expf(-x)); }

// ---- utility kernels ------------------------------------------------------
__global__ void k_zero_u32(uint32_t* p, size_t n) {
  size_t i = (size_t)blockIdx.x * blockDim.x + threadIdx.x;
  if (i < n) p[i] = 0u;
}

// pack f32 [M,K] row-major (row stride lda) into A fragments
__global__ void k_pack_a(const float* __restrict__ src, __bf16* __restrict__ dst,
                         int M, int K, int lda) {
  size_t i = (size_t)blockIdx.x * blockDim.x + threadIdx.x;
  if (i >= (size_t)M * K) return;
  int m = (int)(i / K), k = (int)(i % K);
  dst[a_frag_index(m, k, K >> 5)] = (__bf16)src[(size_t)m * lda + k];
}

// pack weight W f32 [N, lda] rows, columns [0,K) into B fragments (B = W^T)
__global__ void k_pack_b(const float* __restrict__ src, __bf16* __restrict__ dst,
                         int N, int K, int lda) {
  size_t i = (size_t)blockIdx.x * blockDim.x + threadIdx.x;
  if (i >= (size_t)N * K) return;
  int n = (int)(i / K), k = (int)(i % K);
  dst[b_frag_index(n, k, K >> 5)] = (__bf16)src[(size_t)n * lda + k];
}

// encoder embedding gather fused with A-fragment pack: rows m = b*S+s, K=E
__global__ void k_embed_pack_src(const int* __restrict__ tok,
                                 const float* __restrict__ emb,
                                 __bf16* __restrict__ dst, int MR, int E) {
  size_t i = (size_t)blockIdx.x * blockDim.x + threadIdx.x;
  if (i >= (size_t)MR * E) return;
  int m = (int)(i / E), e = (int)(i % E);
  int t = tok[m];
  dst[a_frag_index(m, e, E >> 5)] = (__bf16)emb[(size_t)t * E + e];
}

// decoder embeddings for all steps: dec_in[:,0]=SOS(1), dec_in[:,t]=tgt[:,t]
__global__ void k_embed_pack_dec(const int* __restrict__ tgt,
                                 const float* __restrict__ emb,
                                 __bf16* __restrict__ dst,
                                 int Td, int B, int T, int E) {
  size_t i = (size_t)blockIdx.x * blockDim.x + threadIdx.x;
  if (i >= (size_t)Td * B * E) return;
  int t = (int)(i / ((size_t)B * E));
  int r = (int)(i % ((size_t)B * E));
  int b = r / E, e = r % E;
  int token = (t == 0) ? 1 : tgt[b * T + t];
  dst[(size_t)t * B * E + a_frag_index(b, e, E >> 5)] = (__bf16)emb[(size_t)token * E + e];
}

// ---- WMMA GEMM: C[M,N] = Afrag x Bfrag (+bias[n]) (+Cadd[m*stride+n]) ------
// One wave -> 64x16 macro-tile (4 M-tiles): one B-fragment load is reused by
// 4 back-to-back WMMAs. Mtiles must be a multiple of 4 (true for all GEMMs
// in this model). 4 waves / 128-thread block.
__global__ void k_wmma_gemm(const __bf16* __restrict__ Afrag,
                            const __bf16* __restrict__ Bfrag,
                            float* __restrict__ C,
                            const float* __restrict__ bias,
                            const float* __restrict__ Cadd, long long cadd_stride,
                            int Mtiles, int Ntiles, int KC, int ldc,
                            int beta, int remapB, int remapT) {
  constexpr int MB = 4;
  int wave = blockIdx.x * (blockDim.x >> 5) + (threadIdx.x >> 5);
  int lane = threadIdx.x & 31;
  int Mg = Mtiles / MB;
  if (wave >= Mg * Ntiles) return;              // wave-uniform: EXEC stays full
  int mg = wave % Mg, nt = wave / Mg;
  int mt0 = mg * MB;
  const v16bf* Ap = (const v16bf*)Afrag + (size_t)mt0 * KC * 32 + lane;
  const v16bf* Bp = (const v16bf*)Bfrag + (size_t)nt * KC * 32 + lane;
  v8f acc[MB] = {};
  for (int kt = 0; kt < KC; ++kt) {
    v16bf b = Bp[(size_t)kt * 32];
    if (kt + 1 < KC) {
      __builtin_prefetch(Bp + (size_t)(kt + 1) * 32, 0, 1);
      __builtin_prefetch(Ap + (size_t)(kt + 1) * 32, 0, 1);
    }
#pragma unroll
    for (int i = 0; i < MB; ++i) {
      v16bf a = Ap[(size_t)(i * KC + kt) * 32];
      acc[i] = __builtin_amdgcn_wmma_f32_16x16x32_bf16(
          /*neg_a=*/false, a, /*neg_b=*/false, b,
          /*c_mod=*/(short)0, acc[i], /*reuse_a=*/false, /*reuse_b=*/false);
    }
  }
  int col = nt * 16 + (lane & 15);
  float bv = bias ? bias[col] : 0.0f;
#pragma unroll
  for (int i = 0; i < MB; ++i) {
    int rbase = (mt0 + i) * 16 + ((lane >= 16) ? 8 : 0);
#pragma unroll
    for (int r = 0; r < 8; ++r) {
      int row = rbase + r;
      float v = acc[i][r] + bv;
      if (Cadd) v += Cadd[(size_t)row * cadd_stride + col];
      size_t orow = (size_t)row;
      if (remapB > 0) orow = (size_t)(row % remapB) * remapT + (row / remapB);
      size_t o = orow * (size_t)ldc + col;
      if (beta) C[o] += v; else C[o] = v;
    }
  }
}

// ---- LSTM cell (torch gate order i,f,g,o); writes h, c, h-fragments, h_seq -
__global__ void k_lstm_cell(const float* __restrict__ gates,   // [B,4H]
                            float* __restrict__ h, float* __restrict__ c,
                            __bf16* __restrict__ hfrag,        // A frags, KC=H/32
                            float* __restrict__ h_seq, long long seq_stride,
                            int B, int H) {
  int i = blockIdx.x * blockDim.x + threadIdx.x;
  if (i >= B * H) return;
  int b = i / H, hh = i % H;
  const float* g = gates + (size_t)b * 4 * H;
  float gi = g[hh], gf = g[H + hh], gg = g[2 * H + hh], go = g[3 * H + hh];
  float cn = sigm(gf) * c[i] + sigm(gi) * tanhf(gg);
  float hn = sigm(go) * tanhf(cn);
  c[i] = cn; h[i] = hn;
  hfrag[a_frag_index(b, hh, H >> 5)] = (__bf16)hn;
  if (h_seq) h_seq[(size_t)b * seq_stride + hh] = hn;
}

// ---- Bahdanau attention step: scores->softmax->context (ctx in A-frag form)
__global__ void k_attention(const float* __restrict__ q,       // [B,H]
                            const float* __restrict__ keys,    // [B*S,H]
                            const float* __restrict__ enc,     // [B*S,H]
                            const float* __restrict__ vw, const float* __restrict__ vb,
                            const int* __restrict__ src,       // [B,S]
                            float* __restrict__ attn,          // [B,Td,S]
                            __bf16* __restrict__ ctx_frag,     // [B,H] A frags
                            int t, int Td, int S, int H) {
  __shared__ float qsh[512];
  __shared__ float red[64];
  __shared__ float sm[2];
  int b = blockIdx.x, sI = threadIdx.x;            // 64 threads = 2 waves
  for (int hh = sI; hh < H; hh += 64) qsh[hh] = q[(size_t)b * H + hh];
  __syncthreads();
  const float* kp = keys + ((size_t)b * S + sI) * H;
  float sc = vb[0];
  for (int hh = 0; hh < H; ++hh) sc += tanhf(qsh[hh] + kp[hh]) * vw[hh];
  if (src[b * S + sI] == 0) sc = -1e9f;            // PAD mask
  red[sI] = sc; __syncthreads();
  if (sI == 0) { float m = red[0]; for (int j = 1; j < 64; ++j) m = fmaxf(m, red[j]); sm[0] = m; }
  __syncthreads();
  float e = __expf(sc - sm[0]);
  red[sI] = e; __syncthreads();
  if (sI == 0) { float s = 0.f; for (int j = 0; j < 64; ++j) s += red[j]; sm[1] = s; }
  __syncthreads();
  float w = e / sm[1];
  attn[((size_t)b * Td + t) * S + sI] = w;
  red[sI] = w; __syncthreads();
  for (int j = 0; j < 8; ++j) {
    int hh = sI * 8 + j;
    float a = 0.f;
    for (int ss = 0; ss < S; ++ss) a += red[ss] * enc[((size_t)b * S + ss) * H + hh];
    ctx_frag[a_frag_index(b, hh, H >> 5)] = (__bf16)a;
  }
}

// ---------------------------------------------------------------------------
extern "C" void kernel_launch(void* const* d_in, const int* in_sizes, int n_in,
                              void* d_out, int out_size, void* d_ws, size_t ws_size,
                              hipStream_t stream) {
  constexpr int V = 20000, E = 256, H = 512, B = 128, S = 64, T = 33, Td = 32, G = 4 * H;
  const int*   src     = (const int*)  d_in[0];
  const int*   tgt     = (const int*)  d_in[1];
  const float* enc_emb = (const float*)d_in[2];
  const float* enc_Wih = (const float*)d_in[3];
  const float* enc_Whh = (const float*)d_in[4];
  const float* enc_b   = (const float*)d_in[5];
  const float* dec_emb = (const float*)d_in[6];
  const float* Wq      = (const float*)d_in[7];
  const float* bq      = (const float*)d_in[8];
  const float* Wk      = (const float*)d_in[9];
  const float* bk      = (const float*)d_in[10];
  const float* v_w     = (const float*)d_in[11];
  const float* v_b     = (const float*)d_in[12];
  const float* dec_Wih = (const float*)d_in[13];
  const float* dec_Whh = (const float*)d_in[14];
  const float* dec_b   = (const float*)d_in[15];
  const float* Wo      = (const float*)d_in[16];
  const float* bo      = (const float*)d_in[17];
  float* out_logits = (float*)d_out;
  float* out_attn   = out_logits + (size_t)B * Td * V;

  // ---- workspace layout ----
  char* ws = (char*)d_ws;
  size_t off = 0;
  auto alloc = [&](size_t bytes) { size_t o = off; off = (off + bytes + 255) & ~(size_t)255; return o; };
  size_t o_h      = alloc((size_t)B * H * 4);        // h state   (zeroed; contiguous
  size_t o_c      = alloc((size_t)B * H * 4);        // c state    with c,hfrag)
  size_t o_hfrag  = alloc((size_t)B * H * 2);        // h A-frags
  size_t o_xfrag  = alloc((size_t)B * S * E * 2);
  size_t o_eWih   = alloc((size_t)G * E * 2);
  size_t o_eWhh   = alloc((size_t)G * H * 2);
  size_t o_Wq     = alloc((size_t)H * H * 2);
  size_t o_Wk     = alloc((size_t)H * H * 2);
  size_t o_dWihA  = alloc((size_t)G * E * 2);
  size_t o_dWihB  = alloc((size_t)G * H * 2);
  size_t o_dWhh   = alloc((size_t)G * H * 2);
  size_t o_Wo     = alloc((size_t)V * H * 2);
  size_t o_demb   = alloc((size_t)Td * B * E * 2);
  size_t o_ctx    = alloc((size_t)B * H * 2);
  size_t o_eofrag = alloc((size_t)B * S * H * 2);
  size_t o_Hsfrag = alloc((size_t)Td * B * H * 2);
  size_t o_xg     = alloc((size_t)B * S * G * 4);
  size_t o_gates  = alloc((size_t)B * G * 4);
  size_t o_encout = alloc((size_t)B * S * H * 4);
  size_t o_keys   = alloc((size_t)B * S * H * 4);
  size_t o_q      = alloc((size_t)B * H * 4);
  size_t o_Hs     = alloc((size_t)Td * B * H * 4);

  float*  h      = (float*)(ws + o_h);
  float*  c      = (float*)(ws + o_c);
  __bf16* hfrag  = (__bf16*)(ws + o_hfrag);
  __bf16* xfrag  = (__bf16*)(ws + o_xfrag);
  __bf16* eWih   = (__bf16*)(ws + o_eWih);
  __bf16* eWhh   = (__bf16*)(ws + o_eWhh);
  __bf16* fWq    = (__bf16*)(ws + o_Wq);
  __bf16* fWk    = (__bf16*)(ws + o_Wk);
  __bf16* dWihA  = (__bf16*)(ws + o_dWihA);
  __bf16* dWihB  = (__bf16*)(ws + o_dWihB);
  __bf16* dWhh   = (__bf16*)(ws + o_dWhh);
  __bf16* fWo    = (__bf16*)(ws + o_Wo);
  __bf16* dembf  = (__bf16*)(ws + o_demb);
  __bf16* ctxf   = (__bf16*)(ws + o_ctx);
  __bf16* eofrag = (__bf16*)(ws + o_eofrag);
  __bf16* Hsfrag = (__bf16*)(ws + o_Hsfrag);
  float*  xg     = (float*)(ws + o_xg);
  float*  gates  = (float*)(ws + o_gates);
  float*  encout = (float*)(ws + o_encout);
  float*  keys   = (float*)(ws + o_keys);
  float*  q      = (float*)(ws + o_q);
  float*  Hs     = (float*)(ws + o_Hs);

  auto nblk = [](size_t n, int t) { return (unsigned)((n + t - 1) / t); };
  // waves = (Mtiles/4) * Ntiles, 4 waves per block
  auto gemm = [&](const __bf16* A, const __bf16* Bf, float* C, const float* bias,
                  const float* Cadd, long long cs, int Mt, int Nt, int KC,
                  int ldc, int beta, int rB, int rT) {
    size_t waves = (size_t)(Mt / 4) * Nt;
    unsigned blocks = (unsigned)((waves + 3) / 4);
    k_wmma_gemm<<<blocks, 128, 0, stream>>>(A, Bf, C, bias, Cadd, cs,
                                            Mt, Nt, KC, ldc, beta, rB, rT);
  };

  // zero h, c, hfrag (laid out contiguously)
  size_t zwords = ((size_t)B * H * 4 * 2 + (size_t)B * H * 2) / 4;
  k_zero_u32<<<nblk(zwords, 256), 256, 0, stream>>>((uint32_t*)(ws + o_h), zwords);

  // pack all weight matrices (B fragments, W^T)
  k_pack_b<<<nblk((size_t)G * E, 256), 256, 0, stream>>>(enc_Wih, eWih, G, E, E);
  k_pack_b<<<nblk((size_t)G * H, 256), 256, 0, stream>>>(enc_Whh, eWhh, G, H, H);
  k_pack_b<<<nblk((size_t)H * H, 256), 256, 0, stream>>>(Wq, fWq, H, H, H);
  k_pack_b<<<nblk((size_t)H * H, 256), 256, 0, stream>>>(Wk, fWk, H, H, H);
  k_pack_b<<<nblk((size_t)G * E, 256), 256, 0, stream>>>(dec_Wih,     dWihA, G, E, E + H);
  k_pack_b<<<nblk((size_t)G * H, 256), 256, 0, stream>>>(dec_Wih + E, dWihB, G, H, E + H);
  k_pack_b<<<nblk((size_t)G * H, 256), 256, 0, stream>>>(dec_Whh, dWhh, G, H, H);
  k_pack_b<<<nblk((size_t)V * H, 256), 256, 0, stream>>>(Wo, fWo, V, H, H);

  // ---- encoder ----
  k_embed_pack_src<<<nblk((size_t)B * S * E, 256), 256, 0, stream>>>(src, enc_emb, xfrag, B * S, E);
  // xg[b*S+s, :] = x . Wih^T + enc_b    (M=8192, N=2048, K=256)
  gemm(xfrag, eWih, xg, enc_b, nullptr, 0, (B * S) / 16, G / 16, E / 32, G, 0, 0, 0);
  for (int t = 0; t < S; ++t) {
    // gates = h . Whh^T + xg[:, t, :]
    gemm(hfrag, eWhh, gates, nullptr, xg + (size_t)t * G, (long long)S * G,
         B / 16, G / 16, H / 32, G, 0, 0, 0);
    k_lstm_cell<<<nblk((size_t)B * H, 256), 256, 0, stream>>>(
        gates, h, c, hfrag, encout + (size_t)t * H, (long long)S * H, B, H);
  }
  // keys = enc_out . Wk^T + bk    (M=8192, N=512, K=512)
  k_pack_a<<<nblk((size_t)B * S * H, 256), 256, 0, stream>>>(encout, eofrag, B * S, H, H);
  gemm(eofrag, fWk, keys, bk, nullptr, 0, (B * S) / 16, H / 16, H / 32, H, 0, 0, 0);

  // ---- decoder ----
  k_embed_pack_dec<<<nblk((size_t)Td * B * E, 256), 256, 0, stream>>>(tgt, dec_emb, dembf, Td, B, T, E);
  for (int t = 0; t < Td; ++t) {
    // q = h . Wq^T + bq
    gemm(hfrag, fWq, q, bq, nullptr, 0, B / 16, H / 16, H / 32, H, 0, 0, 0);
    k_attention<<<B, 64, 0, stream>>>(q, keys, encout, v_w, v_b, src,
                                      out_attn, ctxf, t, Td, S, H);
    // gates = emb_t . WihA^T + dec_b ; += ctx . WihB^T ; += h . Whh^T
    gemm(dembf + (size_t)t * B * E, dWihA, gates, dec_b, nullptr, 0,
         B / 16, G / 16, E / 32, G, 0, 0, 0);
    gemm(ctxf, dWihB, gates, nullptr, nullptr, 0, B / 16, G / 16, H / 32, G, 1, 0, 0);
    gemm(hfrag, dWhh, gates, nullptr, nullptr, 0, B / 16, G / 16, H / 32, G, 1, 0, 0);
    k_lstm_cell<<<nblk((size_t)B * H, 256), 256, 0, stream>>>(
        gates, h, c, hfrag, Hs + (size_t)t * B * H, (long long)H, B, H);
  }

  // ---- output projection (hoisted out of scan): logits = Hs . Wo^T + bo ----
  // M = Td*B = 4096 (row = t*B+b), N = 20000, K = 512; row-remap -> [B,Td,V]
  k_pack_a<<<nblk((size_t)Td * B * H, 256), 256, 0, stream>>>(Hs, Hsfrag, Td * B, H, H);
  gemm(Hsfrag, fWo, out_logits, bo, nullptr, 0, (Td * B) / 16, V / 16, H / 32, V,
       0, /*remapB=*/B, /*remapT=*/Td);
}
